// VisionVQVAE_77721728189123
// MI455X (gfx1250) — matrix operand order repaired and checked by reading, not compiled, not run on hardware
//
#include <hip/hip_runtime.h>
#include <hip/hip_bf16.h>

typedef __attribute__((ext_vector_type(16))) _Float16 v16h;
typedef __attribute__((ext_vector_type(8)))  float    v8f;
typedef __attribute__((ext_vector_type(4)))  unsigned v4u;
typedef __attribute__((ext_vector_type(4)))  int      v4i;
typedef __attribute__((ext_vector_type(8)))  int      v8i;

// ---------------------------------------------------------------------------
// f32 -> f16 convert (weights)
// ---------------------------------------------------------------------------
__global__ void cvt_f32_f16(const float* __restrict__ src, _Float16* __restrict__ dst, int n) {
    int i = blockIdx.x * blockDim.x + threadIdx.x;
    if (i < n) dst[i] = (_Float16)src[i];
}

// ---------------------------------------------------------------------------
// Codebook: f32 -> f16 copy + per-row squared norm. One wave per row (D=64).
// ---------------------------------------------------------------------------
__global__ void codebook_prep(const float* __restrict__ cb, _Float16* __restrict__ cbh,
                              float* __restrict__ cn) {
    const int lane = threadIdx.x & 31;
    const int wid  = threadIdx.x >> 5;
    const int row  = blockIdx.x * 8 + wid;           // blockDim = 256 -> 8 rows/block
    const float* src = cb + (size_t)row * 64;
    float a = src[lane];
    float b = src[lane + 32];
    cbh[(size_t)row * 64 + lane]      = (_Float16)a;
    cbh[(size_t)row * 64 + lane + 32] = (_Float16)b;
    float s = a * a + b * b;
    #pragma unroll
    for (int o = 16; o > 0; o >>= 1) s += __shfl_xor(s, o, 32);
    if (lane == 0) cn[row] = s;
}

// ---------------------------------------------------------------------------
// conv1: 3->32, 4x4, s2, p1, fp32 direct (K=48: too small for WMMA to pay off).
// Output stored as f16 NCHW [16][32][128][128] (feeds conv2's WMMA A-gather).
// ---------------------------------------------------------------------------
__global__ void conv1_kernel(const float* __restrict__ x, const float* __restrict__ w1,
                             const float* __restrict__ b1, _Float16* __restrict__ h1) {
    int id = blockIdx.x * blockDim.x + threadIdx.x;  // 16*32*128*128 total
    int ow = id & 127, oh = (id >> 7) & 127, oc = (id >> 14) & 31, b = id >> 19;
    float acc = b1[oc];
    #pragma unroll
    for (int ic = 0; ic < 3; ic++) {
        #pragma unroll
        for (int kh = 0; kh < 4; kh++) {
            int ih = oh * 2 - 1 + kh;
            if ((unsigned)ih < 256u) {
                #pragma unroll
                for (int kw = 0; kw < 4; kw++) {
                    int iw = ow * 2 - 1 + kw;
                    if ((unsigned)iw < 256u)
                        acc = fmaf(x[(size_t)((b * 3 + ic) << 16) + (ih << 8) + iw],
                                   w1[((oc * 3 + ic) << 4) + (kh << 2) + kw], acc);
                }
            }
        }
    }
    h1[id] = (_Float16)fmaxf(acc, 0.0f);   // id is exactly NCHW flat index
}

// ---------------------------------------------------------------------------
// Implicit-GEMM conv via v_wmma_f32_16x16x32_f16.
//   M = 16*OH*OH output pixels, N = 64 output channels, K = IC*16.
// One block = 4 waves; wave w owns N-tile w. Grid = M/16 tiles.
// A fragment layout (16-bit A 16x32, ISA 7.12.2):
//   lanes 0-15: K = {0..7, 16..23}; lanes 16-31: K = {8..15, 24..31} (row = lane&15)
// B fragment layout (16-bit B 32x16): lanes 0-15: K = {0..7, 16..23}; lanes 16-31: K = {8..15, 24..31},
//   column n = lane&15 (weights pre-packed row-major [N][K], K = ic*16+kh*4+kw).
// ---------------------------------------------------------------------------
template<int IC, int IH, int OH, int KTOT, bool OUT_FLAT>
__global__ __launch_bounds__(128) void conv_wmma(const _Float16* __restrict__ in,
                                                 const _Float16* __restrict__ wgt,
                                                 const float* __restrict__ bias,
                                                 _Float16* __restrict__ out) {
    const int lane  = threadIdx.x & 31;
    const int wid   = threadIdx.x >> 5;       // N-tile 0..3
    const int mt    = blockIdx.x;
    const int mrow  = lane & 15;
    const int m     = mt * 16 + mrow;
    const int plane = OH * OH;
    const int b     = m / plane;
    const int rem   = m % plane;
    const int oh    = rem / OH, ow = rem % OH;
    const int kbA   = (lane >> 4) * 8;
    const int kbB   = (lane >> 4) * 8;
    const int ncol  = lane & 15;
    const int n     = wid * 16 + ncol;
    const _Float16* wrow = wgt + (size_t)n * KTOT + kbB;

    v8f acc = {};
    for (int k0 = 0; k0 < KTOT; k0 += 32) {
        v16h a;
        #pragma unroll
        for (int e = 0; e < 16; e++) {
            int k  = k0 + kbA + (e < 8 ? e : 8 + e);
            int ic = k >> 4, kh = (k >> 2) & 3, kw = k & 3;
            int ih = oh * 2 - 1 + kh, iw = ow * 2 - 1 + kw;
            _Float16 v = (_Float16)0.0f;
            if ((unsigned)ih < (unsigned)IH && (unsigned)iw < (unsigned)IH)
                v = in[(size_t)(b * IC + ic) * (IH * IH) + ih * IH + iw];
            a[e] = v;
        }
        v16h bf;
        ((float4*)&bf)[0] = *(const float4*)(wrow + k0);
        ((float4*)&bf)[1] = *(const float4*)(wrow + k0 + 16);
        acc = __builtin_amdgcn_wmma_f32_16x16x32_f16(false, a, false, bf,
                                                     (short)0, acc, false, false);
    }
    #pragma unroll
    for (int r = 0; r < 8; r++) {
        int   m2 = mt * 16 + r + ((lane >> 4) << 3);
        float v  = fmaxf(acc[r] + bias[n], 0.0f);
        if (OUT_FLAT) {
            out[(size_t)m2 * 64 + n] = (_Float16)v;          // [M][64] row-major
        } else {
            int b2 = m2 / plane, r2 = m2 % plane;
            out[(size_t)(b2 * 64 + n) * plane + r2] = (_Float16)v;  // NCHW
        }
    }
}

// ---------------------------------------------------------------------------
// Tensor Data Mover: 1-D chunk copy global -> LDS (D# per ISA ch.8).
// data_size=3 (8 bytes/elem), tile_dim0 = n8 elements, type=2 ("image").
// ---------------------------------------------------------------------------
__device__ __forceinline__ void tdm_load_chunk(const _Float16* gsrc, unsigned lds_off,
                                               unsigned n8, unsigned total8) {
    unsigned long long ga = (unsigned long long)(uintptr_t)gsrc;
    v4u g0;
    g0[0] = 1u;                                            // count=1, user mode
    g0[1] = lds_off;                                       // lds_addr (bytes)
    g0[2] = (unsigned)(ga & 0xFFFFFFFFu);                  // global_addr[31:0]
    g0[3] = (unsigned)((ga >> 32) & 0x1FFFFFFu) | (2u << 30); // addr[56:32] | type=2
    v8i g1;
    g1[0] = (int)(3u << 16);                 // workgroup_mask=0, data_size=3 (8B)
    g1[1] = (int)((total8 & 0xFFFFu) << 16); // tensor_dim0[15:0]
    g1[2] = (int)(total8 >> 16);             // tensor_dim0[31:16], tensor_dim1=0
    g1[3] = (int)(n8 << 16);                 // tile_dim0
    g1[4] = 0;                               // tile_dim1=0 (1-D), tile_dim2=0
    g1[5] = (int)total8;                     // tensor_dim0_stride lo (unused for 1-D)
    g1[6] = 0;
    g1[7] = 0;
    v4i z = {0, 0, 0, 0};
#if defined(__clang_major__) && (__clang_major__ >= 23)
    v8i z8 = {0, 0, 0, 0, 0, 0, 0, 0};
    __builtin_amdgcn_tensor_load_to_lds(g0, g1, z, z, z8, 0);
#else
    __builtin_amdgcn_tensor_load_to_lds(g0, g1, z, z, 0);
#endif
}

// ---------------------------------------------------------------------------
// VQ argmin, TDM-staged + M-batched + packed-index min:
//   - block owns 64 latent rows (4 M-tiles); 8 waves split codebook N-tiles
//   - codebook streamed in 32KB chunks (256 codes) via tensor_load_to_lds,
//     double-buffered in LDS, s_wait_tensorcnt + barrier pipelining
//   - score = cnorm - 2*dot; candidate index packed into low 13 mantissa bits
//     so running argmin is a single v_min_f32 per element
// ---------------------------------------------------------------------------
#define VQ_CHUNK_CODES 256
#define VQ_NCHUNK      32          // 8192 / 256
__global__ __launch_bounds__(256) void vq_argmin(const _Float16* __restrict__ flat,
                                                 const _Float16* __restrict__ cbh,
                                                 const float* __restrict__ cn,
                                                 float* __restrict__ out) {
    __shared__ __align__(16) _Float16 sbuf[2][VQ_CHUNK_CODES * 64];  // 2 x 32 KB
    __shared__ float sred[8][64];

    const int lane = threadIdx.x & 31;
    const int wid  = threadIdx.x >> 5;       // 0..7
    const int ncol = lane & 15;
    const int kbA  = (lane >> 4) * 8;
    const int kbB  = (lane >> 4) * 8;
    const int rowbase = blockIdx.x * 64;

    // A fragments for 4 M-tiles x (K=0..31, K=32..63)
    v16h a0[4], a1[4];
    #pragma unroll
    for (int j = 0; j < 4; j++) {
        const _Float16* arow = flat + (size_t)(rowbase + j * 16 + ncol) * 64;
        ((float4*)&a0[j])[0] = *(const float4*)(arow + kbA);
        ((float4*)&a0[j])[1] = *(const float4*)(arow + kbA + 16);
        ((float4*)&a1[j])[0] = *(const float4*)(arow + 32 + kbA);
        ((float4*)&a1[j])[1] = *(const float4*)(arow + 32 + kbA + 16);
    }

    float mv[4][8];
    #pragma unroll
    for (int j = 0; j < 4; j++)
        #pragma unroll
        for (int r = 0; r < 8; r++) mv[j][r] = __builtin_inff();

    const unsigned lds0 = (unsigned)(uintptr_t)&sbuf[0][0];
    const unsigned lds1 = (unsigned)(uintptr_t)&sbuf[1][0];
    const unsigned total8 = 8192u * 64u * 2u / 8u;   // codebook in 8B units

    if (wid == 0)
        tdm_load_chunk(cbh, lds0, VQ_CHUNK_CODES * 64 * 2 / 8, total8);

    for (int c = 0; c < VQ_NCHUNK; ++c) {
        if (wid == 0) __builtin_amdgcn_s_wait_tensorcnt(0);
        __syncthreads();                                  // chunk c resident
        if (wid == 0 && (c + 1) < VQ_NCHUNK)              // prefetch c+1 (other buffer)
            tdm_load_chunk(cbh + (size_t)(c + 1) * VQ_CHUNK_CODES * 64,
                           ((c + 1) & 1) ? lds1 : lds0,
                           VQ_CHUNK_CODES * 64 * 2 / 8, total8);

        const int bsel = c & 1;
        #pragma unroll
        for (int t = 0; t < 2; ++t) {
            const int ntl = wid + t * 8;                  // local N-tile 0..15
            const _Float16* brow = &sbuf[bsel][(ntl * 16 + ncol) * 64 + kbB];
            v16h b0, b1;
            ((float4*)&b0)[0] = *(const float4*)(brow);
            ((float4*)&b0)[1] = *(const float4*)(brow + 16);
            ((float4*)&b1)[0] = *(const float4*)(brow + 32);
            ((float4*)&b1)[1] = *(const float4*)(brow + 48);
            const int   nid = c * VQ_CHUNK_CODES + ntl * 16 + ncol;
            const float cnv = cn[nid];
            #pragma unroll
            for (int j = 0; j < 4; ++j) {
                v8f acc = {};
                acc = __builtin_amdgcn_wmma_f32_16x16x32_f16(false, a0[j], false, b0,
                                                             (short)0, acc, false, false);
                acc = __builtin_amdgcn_wmma_f32_16x16x32_f16(false, a1[j], false, b1,
                                                             (short)0, acc, false, false);
                #pragma unroll
                for (int r = 0; r < 8; ++r) {
                    float s = fmaf(-2.0f, acc[r], cnv);
                    float p = __int_as_float((__float_as_int(s) & 0xFFFFE000) | nid);
                    mv[j][r] = fminf(mv[j][r], p);
                }
            }
        }
    }

    // min across the 16 lanes sharing each output row (pure float-min tree)
    #pragma unroll
    for (int off = 1; off < 16; off <<= 1)
        #pragma unroll
        for (int j = 0; j < 4; j++)
            #pragma unroll
            for (int r = 0; r < 8; r++)
                mv[j][r] = fminf(mv[j][r], __shfl_xor(mv[j][r], off, 32));

    if (ncol == 0) {
        #pragma unroll
        for (int j = 0; j < 4; j++)
            #pragma unroll
            for (int r = 0; r < 8; r++)
                sred[wid][j * 16 + r + ((lane >> 4) << 3)] = mv[j][r];
    }
    __syncthreads();
    if (threadIdx.x < 64) {
        float b = sred[0][threadIdx.x];
        #pragma unroll
        for (int w = 1; w < 8; w++) b = fminf(b, sred[w][threadIdx.x]);
        out[rowbase + threadIdx.x] = (float)(__float_as_int(b) & 0x1FFF);
    }
    if (blockIdx.x == 0 && threadIdx.x == 0) out[16384] = 0.0f;  // loss term
}

// ---------------------------------------------------------------------------
extern "C" void kernel_launch(void* const* d_in, const int* in_sizes, int n_in,
                              void* d_out, int out_size, void* d_ws, size_t ws_size,
                              hipStream_t stream) {
    const float* x  = (const float*)d_in[0];
    const float* w1 = (const float*)d_in[1];
    const float* b1 = (const float*)d_in[2];
    const float* w2 = (const float*)d_in[3];
    const float* b2 = (const float*)d_in[4];
    const float* w3 = (const float*)d_in[5];
    const float* b3 = (const float*)d_in[6];
    const float* cb = (const float*)d_in[7];
    float* out = (float*)d_out;
    (void)in_sizes; (void)n_in; (void)out_size; (void)ws_size;

    char* ws = (char*)d_ws;
    size_t off = 0;
    auto take = [&](size_t bytes) -> char* {
        char* p = ws + off;
        off = (off + bytes + 255) & ~(size_t)255;
        return p;
    };
    _Float16* cbh = (_Float16*)take((size_t)8192 * 64 * 2);          // 1 MB
    float*    cnm = (float*)   take((size_t)8192 * 4);               // 32 KB
    _Float16* w2h = (_Float16*)take((size_t)64 * 512 * 2);           // 64 KB
    _Float16* w3h = (_Float16*)take((size_t)64 * 1024 * 2);          // 128 KB
    _Float16* h1  = (_Float16*)take((size_t)16 * 32 * 128 * 128 * 2);// 16.8 MB
    _Float16* h2  = (_Float16*)take((size_t)16 * 64 * 64 * 64 * 2);  // 8.4 MB
    _Float16* fl  = (_Float16*)take((size_t)16384 * 64 * 2);         // 2 MB

    cvt_f32_f16<<<128, 256, 0, stream>>>(w2, w2h, 64 * 512);
    cvt_f32_f16<<<256, 256, 0, stream>>>(w3, w3h, 64 * 1024);
    codebook_prep<<<1024, 256, 0, stream>>>(cb, cbh, cnm);
    conv1_kernel<<<32768, 256, 0, stream>>>(x, w1, b1, h1);
    // conv2: IC=32, 128->64, K=512; output NCHW f16
    conv_wmma<32, 128, 64, 512, false><<<4096, 128, 0, stream>>>(h1, w2h, b2, h2);
    // conv3: IC=64, 64->32, K=1024; output row-major [16384][64] f16 (= flat)
    conv_wmma<64, 64, 32, 1024, true><<<1024, 128, 0, stream>>>(h2, w3h, b3, fl);
    vq_argmin<<<256, 256, 0, stream>>>(fl, cbh, cnm, out);
}
